// WindowAttention_53867479826887
// MI455X (gfx1250) — matrix operand order repaired and checked by reading, not compiled, not run on hardware
//
#include <hip/hip_runtime.h>
#include <hip/hip_bf16.h>

typedef __attribute__((ext_vector_type(16))) __bf16 v16bf;
typedef __attribute__((ext_vector_type(8)))  float  v8f;

union Frag {
    v16bf v;
    uint4 q[2];
    unsigned short us[16];
};

static __device__ __forceinline__ unsigned short f2bf(float f) {
    unsigned u = __float_as_uint(f);
    u += 0x7fffu + ((u >> 16) & 1u);            // round-to-nearest-even
    return (unsigned short)(u >> 16);
}

// ---------------------------------------------------------------------------
// Tiled WMMA GEMM: C[M,NMAT] = A[M,512] * B[512,NMAT] + bias
//   Block tile 128x128, K-step 32, 256 threads = 8 waves, wave tile 32x64.
//   Software pipelined: tile k+1 fetched to registers while tile k computes.
// ---------------------------------------------------------------------------
template<bool A_BF16, bool OUT_BF16, int NMAT>
__global__ __launch_bounds__(256)
void wmma_gemm(const void* __restrict__ Av, const float* __restrict__ Bw,
               const float* __restrict__ bias, void* __restrict__ Outv)
{
    constexpr int K = 512;
    constexpr int KT = K / 32;                   // 16 k-steps
    constexpr int PITCH = 40;                    // shorts; 80B rows, conflict-free
    __shared__ __align__(16) unsigned short As[128 * PITCH];
    __shared__ __align__(16) unsigned short Bs[128 * PITCH];

    const int t    = threadIdx.x;
    const int lane = t & 31;
    const int wid  = t >> 5;
    const int wm   = wid & 3;                    // M sub-block (4 x 32 rows)
    const int wn   = wid >> 2;                   // N sub-block (2 x 64 cols)
    const int m0   = blockIdx.y * 128;
    const int n0   = blockIdx.x * 128;
    const int g    = lane >> 4;                  // half-wave
    const int l    = lane & 15;

    // staging assignments
    const int ar  = t >> 1;                      // A: row, half-k per thread
    const int akh = (t & 1) * 16;
    const int bn  = t & 127;                     // B: column, half-k per thread
    const int bkb = (t >> 7) * 16;

    const unsigned short* Abf = (const unsigned short*)Av;
    const float*          Af  = (const float*)Av;

    // staging registers (raw; converted at store time)
    uint4  aq[2];                                // A_BF16 path
    float4 af[4];                                // fp32 path
    float  bf[16];

    auto fetch = [&](int k0) {
        if (A_BF16) {
            const uint4* src = (const uint4*)(Abf + (size_t)(m0 + ar) * K + k0 + akh);
            aq[0] = src[0];
            aq[1] = src[1];
        } else {
            const float4* src = (const float4*)(Af + (size_t)(m0 + ar) * K + k0 + akh);
            #pragma unroll
            for (int i = 0; i < 4; ++i) af[i] = src[i];
        }
        const float* src = Bw + (size_t)(k0 + bkb) * NMAT + n0 + bn;
        #pragma unroll
        for (int i = 0; i < 16; ++i) bf[i] = src[(size_t)i * NMAT];
    };

    v8f acc[2][4];
    #pragma unroll
    for (int mi = 0; mi < 2; ++mi)
        #pragma unroll
        for (int ni = 0; ni < 4; ++ni)
            acc[mi][ni] = (v8f){0.f,0.f,0.f,0.f,0.f,0.f,0.f,0.f};

    fetch(0);

    for (int kt = 0; kt < KT; ++kt) {
        __syncthreads();                         // previous tile fully consumed
        // ---- commit staged registers to LDS (bf16) ----
        {
            unsigned short* dst = &As[ar * PITCH + akh];
            if (A_BF16) {
                ((uint4*)dst)[0] = aq[0];
                ((uint4*)dst)[1] = aq[1];
            } else {
                union { uint4 q[2]; unsigned short us[16]; } u;
                #pragma unroll
                for (int i = 0; i < 4; ++i) {
                    u.us[i * 4 + 0] = f2bf(af[i].x);
                    u.us[i * 4 + 1] = f2bf(af[i].y);
                    u.us[i * 4 + 2] = f2bf(af[i].z);
                    u.us[i * 4 + 3] = f2bf(af[i].w);
                }
                ((uint4*)dst)[0] = u.q[0];
                ((uint4*)dst)[1] = u.q[1];
            }
            union { uint4 q[2]; unsigned short us[16]; } u;
            #pragma unroll
            for (int i = 0; i < 16; ++i) u.us[i] = f2bf(bf[i]);
            unsigned short* db = &Bs[bn * PITCH + bkb];
            ((uint4*)db)[0] = u.q[0];
            ((uint4*)db)[1] = u.q[1];
        }
        __syncthreads();

        // ---- prefetch next tile while WMMAs run ----
        if (kt + 1 < KT) fetch((kt + 1) * 32);

        // ---- fragments ----
        Frag a[2], b[4];
        #pragma unroll
        for (int mi = 0; mi < 2; ++mi) {
            const int base = (32 * wm + 16 * mi + l) * PITCH;
            a[mi].q[0] = *(const uint4*)&As[base + 16 * g];       // k = 16g..16g+7
            a[mi].q[1] = *(const uint4*)&As[base + 8 + 16 * g];   // k = 16g+8..
        }
        #pragma unroll
        for (int ni = 0; ni < 4; ++ni) {
            const int base = (64 * wn + 16 * ni + l) * PITCH + 16 * g; // k 16g..16g+15
            b[ni].q[0] = *(const uint4*)&Bs[base];
            b[ni].q[1] = *(const uint4*)&Bs[base + 8];
        }
        #pragma unroll
        for (int mi = 0; mi < 2; ++mi)
            #pragma unroll
            for (int ni = 0; ni < 4; ++ni)
                acc[mi][ni] = __builtin_amdgcn_wmma_f32_16x16x32_bf16(
                    false, a[mi].v, false, b[ni].v, (short)0, acc[mi][ni], false, false);
    }

    // ---- epilogue: add bias, store ----
    #pragma unroll
    for (int mi = 0; mi < 2; ++mi) {
        const int mbase = m0 + 32 * wm + 16 * mi + 8 * g;
        #pragma unroll
        for (int ni = 0; ni < 4; ++ni) {
            const int col = n0 + 64 * wn + 16 * ni + l;
            const float bv = bias[col];
            #pragma unroll
            for (int r = 0; r < 8; ++r) {
                const float v = acc[mi][ni][r] + bv;
                const size_t off = (size_t)(mbase + r) * NMAT + col;
                if (OUT_BF16) ((unsigned short*)Outv)[off] = f2bf(v);
                else          ((float*)Outv)[off]          = v;
            }
        }
    }
}

// ---------------------------------------------------------------------------
// Window attention core: one wave per (window, head). 8 waves / block.
//   qkv ws: bf16 [B*49, 1536], col = s*512 + h*32 + d
//   out ws: bf16 [B*49, 512],  col = h*32 + d
// NOTE: Q/K fragment loads are issued unguarded for rows 49..63. Those rows
// only feed masked-out S columns / discarded O rows, the clamp in the softmax
// path scrubs any NaN/Inf they could produce, and the addresses stay inside
// d_ws (the attn region follows the qkv region).
// ---------------------------------------------------------------------------
__global__ __launch_bounds__(256)
void attn_kernel(const unsigned short* __restrict__ qkv,
                 const float* __restrict__ bias_table,
                 unsigned short* __restrict__ outws)
{
    constexpr int   VP    = 72;                  // LDS pitch (shorts); 144B rows
    constexpr float SCALE = 0.1767766952966369f; // 32^-0.5
    __shared__ __align__(16) unsigned short Vsh[8 * 32 * VP]; // V^T per wave
    __shared__ __align__(16) unsigned short Psh[8 * 16 * VP]; // P block per wave
    __shared__ float Bsh[8 * 169];                            // bias per head

    const int t    = threadIdx.x;
    const int lane = t & 31;
    const int wid  = t >> 5;
    const int g    = lane >> 4;
    const int l    = lane & 15;
    const int b    = blockIdx.x >> 1;
    const int h    = (blockIdx.x & 1) * 8 + wid;

    const unsigned short* base = qkv + (size_t)b * 49 * 1536 + h * 32;
    const unsigned short* qB = base;
    const unsigned short* kB = base + 512;
    const unsigned short* vB = base + 1024;
    unsigned short* Vw = &Vsh[wid * 32 * VP];
    unsigned short* Pw = &Psh[wid * 16 * VP];
    float*          Bb = &Bsh[wid * 169];

    // relative-position bias for this head -> LDS
    for (int i = lane; i < 169; i += 32) Bb[i] = bias_table[i * 16 + h];

    // stage V transposed: Vw[d*VP + k], zero-pad k in [49,64)
    for (int k = lane; k < 64; k += 32) {
        if (k < 49) {
            const uint4* src = (const uint4*)(vB + (size_t)k * 1536);
            union { uint4 q[4]; unsigned short us[32]; } u;
            u.q[0] = src[0]; u.q[1] = src[1]; u.q[2] = src[2]; u.q[3] = src[3];
            #pragma unroll
            for (int d = 0; d < 32; ++d) Vw[d * VP + k] = u.us[d];
        } else {
            #pragma unroll
            for (int d = 0; d < 32; ++d) Vw[d * VP + k] = 0;
        }
    }

    // K fragments (B layout: lane holds key n = 16j+l, d block 16g..16g+15)
    Frag kf[4];
    #pragma unroll
    for (int j = 0; j < 4; ++j) {
        const uint4* s = (const uint4*)(kB + (size_t)(16 * j + l) * 1536 + 16 * g);
        kf[j].q[0] = s[0]; kf[j].q[1] = s[1];
    }

    // per-lane column constants (clamped for the bias index; mask as addend)
    int in_[4], jn_[4];
    float nmask[4];                              // 0 if col valid, else -2e30
    #pragma unroll
    for (int j = 0; j < 4; ++j) {
        const int n  = 16 * j + l;
        const int nc = (n < 49) ? n : 48;
        nmask[j] = (n < 49) ? 0.f : -2e30f;
        in_[j] = nc / 7;
        jn_[j] = nc - in_[j] * 7;
    }

    asm volatile("s_wait_dscnt 0" ::: "memory");  // Vw / Bb ready (same wave)

    // V fragments: [ks][dt], lane holds column d = 16dt+l, keys 32ks+16g..+15
    Frag vf[2][2];
    #pragma unroll
    for (int ks = 0; ks < 2; ++ks)
        #pragma unroll
        for (int dt = 0; dt < 2; ++dt) {
            const uint4* s = (const uint4*)&Vw[(16 * dt + l) * VP + 32 * ks + 16 * g];
            vf[ks][dt].q[0] = s[0]; vf[ks][dt].q[1] = s[1];
        }

    const v8f vzero = (v8f){0.f,0.f,0.f,0.f,0.f,0.f,0.f,0.f};

    #pragma unroll
    for (int i = 0; i < 4; ++i) {                // 16-row tiles of queries
        // Q fragment (A layout), unguarded (see note above)
        Frag qf;
        {
            const unsigned short* qp = qB + (size_t)(16 * i + l) * 1536;
            qf.q[0] = *(const uint4*)(qp + 16 * g);
            qf.q[1] = *(const uint4*)(qp + 8 + 16 * g);
        }

        // S = Q K^T  (one K-step, D=32)
        v8f s[4];
        #pragma unroll
        for (int j = 0; j < 4; ++j)
            s[j] = __builtin_amdgcn_wmma_f32_16x16x32_bf16(
                false, qf.v, false, kf[j].v, (short)0, vzero, false, false);

        // scale + bias + mask — all unconditional (load feeds fmac directly;
        // clamp scrubs NaN/Inf; masking is a constant-select addend)
        float vals[4][8], rmax[8], rsum[8];
        #pragma unroll
        for (int r = 0; r < 8; ++r) {
            const int  m     = 16 * i + r + 8 * g;
            const bool vm    = (i < 3) || (m < 49);      // folds for i<3
            const int  mc    = vm ? m : 48;
            const float mmsk = vm ? 0.f : -2e30f;
            const int im = mc / 7;
            const int jm = mc - im * 7;
            float mx = -3e30f;
            #pragma unroll
            for (int j = 0; j < 4; ++j) {
                const int idx = (im - in_[j] + 6) * 13 + (jm - jn_[j] + 6);
                float v = s[j][r] * SCALE + Bb[idx];      // always executed
                v = fminf(fmaxf(v, -1e30f), 1e30f);       // NaN/Inf scrub
                v += nmask[j] + mmsk;                     // mask as addend
                vals[j][r] = v;
                mx = fmaxf(mx, v);
            }
            rmax[r] = mx;
        }
        #pragma unroll
        for (int r = 0; r < 8; ++r)
            #pragma unroll
            for (int msk = 1; msk < 16; msk <<= 1)
                rmax[r] = fmaxf(rmax[r], __shfl_xor(rmax[r], msk, 32));
        #pragma unroll
        for (int r = 0; r < 8; ++r) {
            float sm = 0.f;
            #pragma unroll
            for (int j = 0; j < 4; ++j) {
                const float p = __expf(vals[j][r] - rmax[r]);  // masked -> 0
                vals[j][r] = p;
                sm += p;
            }
            rsum[r] = sm;
        }
        #pragma unroll
        for (int r = 0; r < 8; ++r)
            #pragma unroll
            for (int msk = 1; msk < 16; msk <<= 1)
                rsum[r] += __shfl_xor(rsum[r], msk, 32);

        // normalize, P -> LDS (bf16) so it can be reloaded as A fragments
        #pragma unroll
        for (int r = 0; r < 8; ++r) {
            const float inv = __builtin_amdgcn_rcpf(rsum[r]); // rsum >= 1
            #pragma unroll
            for (int j = 0; j < 4; ++j)
                Pw[(r + 8 * g) * VP + 16 * j + l] = f2bf(vals[j][r] * inv);
        }
        asm volatile("s_wait_dscnt 0" ::: "memory");

        // O = P V  (2 K-steps of 32 keys, 2 d-tiles)
        v8f o[2] = {vzero, vzero};
        #pragma unroll
        for (int ks = 0; ks < 2; ++ks) {
            Frag pf;
            const unsigned short* pp = Pw + (size_t)l * VP + 32 * ks;
            pf.q[0] = *(const uint4*)(pp + 16 * g);
            pf.q[1] = *(const uint4*)(pp + 8 + 16 * g);
            #pragma unroll
            for (int dt = 0; dt < 2; ++dt)
                o[dt] = __builtin_amdgcn_wmma_f32_16x16x32_bf16(
                    false, pf.v, false, vf[ks][dt].v, (short)0, o[dt], false, false);
        }
        // store O rows (bf16 ws); guard folds away for i<3
        #pragma unroll
        for (int dt = 0; dt < 2; ++dt)
            #pragma unroll
            for (int r = 0; r < 8; ++r) {
                const int m = 16 * i + r + 8 * g;
                if (i < 3 || m < 49) {
                    const int d = 16 * dt + l;
                    outws[(size_t)(b * 49 + m) * 512 + h * 32 + d] = f2bf(o[dt][r]);
                }
            }
    }
}

// ---------------------------------------------------------------------------
extern "C" void kernel_launch(void* const* d_in, const int* in_sizes, int n_in,
                              void* d_out, int out_size, void* d_ws, size_t ws_size,
                              hipStream_t stream) {
    (void)in_sizes; (void)n_in; (void)out_size; (void)ws_size;
    const float* x          = (const float*)d_in[0]; // [4096,49,512]
    const float* qkv_w      = (const float*)d_in[1]; // [512,1536]
    const float* qkv_b      = (const float*)d_in[2]; // [1536]
    const float* proj_w     = (const float*)d_in[3]; // [512,512]
    const float* proj_b     = (const float*)d_in[4]; // [512]
    const float* bias_table = (const float*)d_in[5]; // [169,16]

    constexpr int Mrows = 4096 * 49;                 // 200704 (= 128 * 1568)
    unsigned short* qkvws = (unsigned short*)d_ws;                    // bf16 [M,1536]
    unsigned short* attws = qkvws + (size_t)Mrows * 1536;             // bf16 [M,512]

    // 1) QKV projection (fp32 in -> bf16 ws)
    wmma_gemm<false, true, 1536><<<dim3(12, 1568), 256, 0, stream>>>(
        x, qkv_w, qkv_b, qkvws);
    // 2) windowed attention per (window, head)
    attn_kernel<<<dim3(4096 * 2), 256, 0, stream>>>(qkvws, bias_table, attws);
    // 3) output projection (bf16 ws -> fp32 out)
    wmma_gemm<true, false, 512><<<dim3(4, 1568), 256, 0, stream>>>(
        attws, proj_w, proj_b, (float*)d_out);
}
